// TactileNetwork_27401891348684
// MI455X (gfx1250) — matrix operand order repaired and checked by reading, not compiled, not run on hardware
//
#include <hip/hip_runtime.h>
#include <hip/hip_bf16.h>

typedef __attribute__((ext_vector_type(16))) _Float16 v16h;
typedef __attribute__((ext_vector_type(8)))  float    v8f;

// A-fragment (16x32 f16, wave32) within-lane K index for element e (ISA 7.12.2):
// lanes 0-15: VGPR0-3 -> K 0..7, VGPR4-7 -> K 16..23 ; lanes 16-31: +8
__device__ __forceinline__ int akidx(int e, int hf) {
    return (e < 8) ? (e + 8 * hf) : (e + 8 + 8 * hf);
}

// ---------------------------------------------------------------------------
// CDNA5 async global->LDS copy (ASYNCcnt-tracked). The builtin's pointer
// params are typed as int4 vectors in addrspace(1)/(3) (per hipcc diagnostic).
// ---------------------------------------------------------------------------
#if __has_builtin(__builtin_amdgcn_global_load_async_to_lds_b128)
#define HAS_ASYNC_LDS 1
#else
#define HAS_ASYNC_LDS 0
#endif

typedef __attribute__((__vector_size__(4 * sizeof(int)))) int v4i_t;

__device__ __forceinline__ void async_copy16(const float* g, float* l) {
#if HAS_ASYNC_LDS
    // generic->addrspace via integer: LDS generic VA keeps the byte offset in
    // the low 32 bits (ISA 10.2 aperture mapping); global VA is the full value.
    __builtin_amdgcn_global_load_async_to_lds_b128(
        (__attribute__((address_space(1))) v4i_t*)(unsigned long long)(const void*)g,
        (__attribute__((address_space(3))) v4i_t*)(unsigned)(unsigned long long)(const void*)l,
        0, 0);
#else
    *(float4*)l = *(const float4*)g;
#endif
}

__device__ __forceinline__ void async_wait_all() {
#if HAS_ASYNC_LDS
#if __has_builtin(__builtin_amdgcn_s_wait_asynccnt)
    __builtin_amdgcn_s_wait_asynccnt(0);
#else
    asm volatile("s_wait_asynccnt 0x0" ::: "memory");
#endif
#endif
}

// ---------------------------------------------------------------------------
// conv1: (256,1,128,128) -> (256,32,64,64), k=4 s=2 p=1, ReLU.  Memory bound;
// direct VALU with weights staged in LDS. One thread per spatial position,
// all 32 output channels (loads each input patch once).
// ---------------------------------------------------------------------------
__global__ __launch_bounds__(256)
void conv1_kernel(const float* __restrict__ x, const float* __restrict__ w1,
                  const float* __restrict__ b1, float* __restrict__ h1) {
    __shared__ float ws[512];
    __shared__ float bs[32];
    int t = threadIdx.x;
    for (int i = t; i < 512; i += 256) ws[i] = w1[i];
    if (t < 32) bs[t] = b1[t];
    __syncthreads();

    int pos = blockIdx.x * 256 + t;          // over B*64*64 = 1048576
    int ox = pos & 63, oy = (pos >> 6) & 63, b = pos >> 12;
    const float* xb = x + (size_t)b * 128 * 128;

    float patch[16];
#pragma unroll
    for (int ky = 0; ky < 4; ++ky) {
        int iy = 2 * oy - 1 + ky;
#pragma unroll
        for (int kx = 0; kx < 4; ++kx) {
            int ix = 2 * ox - 1 + kx;
            patch[ky * 4 + kx] = ((unsigned)iy < 128u && (unsigned)ix < 128u)
                                     ? xb[iy * 128 + ix] : 0.0f;
        }
    }
    float* ob = h1 + (size_t)b * 32 * 4096 + oy * 64 + ox;
#pragma unroll 4
    for (int oc = 0; oc < 32; ++oc) {
        float acc = bs[oc];
#pragma unroll
        for (int k = 0; k < 16; ++k) acc = fmaf(patch[k], ws[oc * 16 + k], acc);
        ob[(size_t)oc * 4096] = fmaxf(acc, 0.0f);
    }
}

// ---------------------------------------------------------------------------
// Implicit-GEMM conv via WMMA f32<=f16xf16: M=COUT channels, K=CIN*16 taps,
// N=B*HWOUT^2 positions.
//  * 8 waves/block own 128 consecutive positions = ROWS full output rows of
//    one image; gridDim.y selects the 16-channel M tile.
//  * The stride-2 halo tile (CIN x 2*ROWS+2 x HWIN, fp32) is staged in LDS:
//    pre-zeroed (halo rows + pad columns), then interior rows streamed in with
//    async global->LDS b128 copies (16B aligned: rows start at +4 floats, the
//    left pad doubles as the ix=-1 halo column). s_wait_asynccnt + barrier.
//  * B fragments are then built from LDS with branch-free addressing and
//    converted fp32->f16 right before v_wmma_f32_16x16x32_f16.
// ---------------------------------------------------------------------------
template <int CIN, int COUT, int HWIN, int HWOUT>
__global__ __launch_bounds__(256)
void conv_wmma_kernel(const float* __restrict__ in, const float* __restrict__ w,
                      const float* __restrict__ bias, float* __restrict__ out) {
    constexpr int K       = CIN * 16;
    constexpr int ROWS    = 128 / HWOUT;      // output rows handled per block
    constexpr int ROWS_IN = 2 * ROWS + 2;     // input halo rows
    constexpr int WPAD    = HWIN + 8;         // 4 left pad, >=2 right pad
    constexpr int TILE    = CIN * ROWS_IN * WPAD;

    extern __shared__ char smem[];
    _Float16* wLds = (_Float16*)smem;                          // 16*K halves
    float*    sIn  = (float*)(smem + (size_t)16 * K * sizeof(_Float16));

    int t = threadIdx.x;
    int lane = t & 31, wave = t >> 5;
    int hf = lane >> 4, nl = lane & 15;
    int mBase = blockIdx.y * 16;

    int nBlock   = blockIdx.x * 128;
    int b        = nBlock / (HWOUT * HWOUT);
    int posInImg = nBlock % (HWOUT * HWOUT);
    int oy0      = posInImg / HWOUT;
    int iyBase   = 2 * oy0 - 1;
    const float* inb = in + (size_t)b * CIN * HWIN * HWIN;

    // Stage weight slab fp32->f16 and zero the halo tile (DS writes).
    for (int i = t; i < 16 * K; i += 256)
        wLds[i] = (_Float16)w[(size_t)(mBase + i / K) * K + (i % K)];
    for (int i = t; i < TILE; i += 256) sIn[i] = 0.0f;
    __syncthreads();   // DS zeroing complete before async writes land

    // Async-copy valid interior rows into cols [4, 4+HWIN).
    constexpr int SEGS   = HWIN / 4;
    constexpr int CHUNKS = CIN * ROWS_IN * SEGS;
    for (int c = t; c < CHUNKS; c += 256) {
        int seg = c % SEGS;
        int row = (c / SEGS) % ROWS_IN;
        int cin = c / (SEGS * ROWS_IN);
        int iy  = iyBase + row;
        if ((unsigned)iy < (unsigned)HWIN) {
            const float* g = inb + (size_t)cin * HWIN * HWIN + iy * HWIN + seg * 4;
            float*       l = sIn + (cin * ROWS_IN + row) * WPAD + 4 + seg * 4;
            async_copy16(g, l);
        }
    }
    async_wait_all();
    __syncthreads();

    int idx = wave * 16 + nl;                 // position within block's 128
    int oyl = idx / HWOUT, ox = idx % HWOUT;  // local output row / column
    int colBase = 2 * ox + 3;                 // (2*ox - 1) + kx + 4 left pad

    v8f c = {};
    for (int kb = 0; kb < K; kb += 32) {
        v16h a, bf;
#pragma unroll
        for (int e = 0; e < 16; ++e)
            a[e] = wLds[nl * K + kb + e + 16 * hf];
#pragma unroll
        for (int e = 0; e < 16; ++e) {
            int k = kb + e + 16 * hf;         // B-frag per-lane K index
            int cin = k >> 4, tap = k & 15;
            int r = 2 * oyl + (tap >> 2);
            bf[e] = (_Float16)sIn[(cin * ROWS_IN + r) * WPAD + colBase + (tap & 3)];
        }
        c = __builtin_amdgcn_wmma_f32_16x16x32_f16(false, a, false, bf,
                                                   (short)0, c, false, false);
    }
#pragma unroll
    for (int r8 = 0; r8 < 8; ++r8) {
        int m = mBase + r8 + 8 * hf;          // C/D layout: M = r + 8*half
        float v = fmaxf(c[r8] + bias[m], 0.0f);
        out[((size_t)b * COUT + m) * (HWOUT * HWOUT) + posInImg + idx] = v;
    }
}

// ---------------------------------------------------------------------------
// Global average pool: h3 (256,128,16,16) -> feat (256,128)
// ---------------------------------------------------------------------------
__global__ __launch_bounds__(256)
void pool_kernel(const float* __restrict__ h3, float* __restrict__ feat) {
    int i = blockIdx.x * 256 + threadIdx.x;           // over 32768 (b,c) rows
    const float4* p = (const float4*)(h3 + (size_t)i * 256);
    float s = 0.0f;
#pragma unroll 8
    for (int j = 0; j < 64; ++j) { float4 v = p[j]; s += v.x + v.y + v.z + v.w; }
    feat[i] = s * (1.0f / 256.0f);
}

// ---------------------------------------------------------------------------
// C[M,N] = A[M,K] @ W[N,K]^T + bias[N]  via WMMA, one wave per 16x16 tile.
// ---------------------------------------------------------------------------
__global__ __launch_bounds__(32)
void gemm_nt_wmma(const float* __restrict__ A, const float* __restrict__ W,
                  const float* __restrict__ bias, float* __restrict__ C,
                  int N, int K) {
    int lane = threadIdx.x;
    int hf = lane >> 4, nl = lane & 15;
    int m0 = blockIdx.y * 16, n0 = blockIdx.x * 16;
    const float* arow = A + (size_t)(m0 + nl) * K;
    const float* wrow = W + (size_t)(n0 + nl) * K;

    v8f c = {};
    for (int kb = 0; kb < K; kb += 32) {
        v16h a, b;
#pragma unroll
        for (int e = 0; e < 16; ++e) a[e] = (_Float16)arow[kb + e + 16 * hf];
#pragma unroll
        for (int e = 0; e < 16; ++e) b[e] = (_Float16)wrow[kb + e + 16 * hf];
        c = __builtin_amdgcn_wmma_f32_16x16x32_f16(false, a, false, b,
                                                   (short)0, c, false, false);
    }
#pragma unroll
    for (int r = 0; r < 8; ++r) {
        int m = m0 + r + 8 * hf, n = n0 + nl;
        C[(size_t)m * N + n] = c[r] + bias[n];
    }
}

// ---------------------------------------------------------------------------
// VQ: per batch row, argmin_k ||z_e - emb[k]||^2 over 512 codes (first-min
// tie-break like jnp.argmin), gather z_q, emit per-row loss partial.
// ---------------------------------------------------------------------------
__global__ __launch_bounds__(256)
void vq_kernel(const float* __restrict__ z_e_all, const float* __restrict__ emb,
               float* __restrict__ z_q_out, float* __restrict__ ploss) {
    __shared__ float ze[256];
    __shared__ float dmin[256];
    __shared__ int   didx[256];
    int b = blockIdx.x, t = threadIdx.x;
    ze[t] = z_e_all[(size_t)b * 256 + t];
    __syncthreads();

    float best = __builtin_inff();
    int bi = 0;
#pragma unroll
    for (int k0 = 0; k0 < 2; ++k0) {
        int k = t * 2 + k0;
        const float* ek = emb + (size_t)k * 256;
        float d = 0.0f;
        for (int e = 0; e < 256; ++e) { float df = ze[e] - ek[e]; d = fmaf(df, df, d); }
        if (d < best) { best = d; bi = k; }
    }
    dmin[t] = best; didx[t] = bi;
    __syncthreads();
    for (int s = 128; s > 0; s >>= 1) {
        if (t < s) {
            if (dmin[t + s] < dmin[t] ||
                (dmin[t + s] == dmin[t] && didx[t + s] < didx[t])) {
                dmin[t] = dmin[t + s]; didx[t] = didx[t + s];
            }
        }
        __syncthreads();
    }
    int bestk = didx[0];
    __syncthreads();
    float zq = emb[(size_t)bestk * 256 + t];
    z_q_out[(size_t)b * 256 + t] = zq;
    float df = zq - ze[t];
    dmin[t] = df * df;
    __syncthreads();
    for (int s = 128; s > 0; s >>= 1) { if (t < s) dmin[t] += dmin[t + s]; __syncthreads(); }
    if (t == 0) ploss[b] = dmin[0];
}

__global__ __launch_bounds__(256)
void loss_reduce(const float* __restrict__ ploss, float* __restrict__ loss_out) {
    __shared__ float s[256];
    int t = threadIdx.x;
    s[t] = ploss[t];
    __syncthreads();
    for (int k = 128; k > 0; k >>= 1) { if (t < k) s[t] += s[t + k]; __syncthreads(); }
    if (t == 0) *loss_out = s[0] * (1.0f / 65536.0f);
}

// ---------------------------------------------------------------------------
// Fused decoder, one block per batch element.
// deconv k=4 s=2 p=1 (lax correlation, OIHW, lhs_dilation=2, pad=2):
//   stage1 (1x1 -> 2x2):  d1[o][y][x] = relu(db1[o] + sum_i lat[i]*dw1[o][i][2-y][2-x])
//   stageN:               ky = 2 + 2*iy - y  (valid iff 0<=ky<4)
// ---------------------------------------------------------------------------
__global__ __launch_bounds__(256)
void decoder_kernel(const float* __restrict__ lat,
                    const float* __restrict__ dw1, const float* __restrict__ db1,
                    const float* __restrict__ dw2, const float* __restrict__ db2,
                    const float* __restrict__ dw3, const float* __restrict__ db3,
                    float* __restrict__ recon) {
    __shared__ float l[128];
    __shared__ float d1s[64 * 4];
    __shared__ float d2s[32 * 16];
    int b = blockIdx.x, t = threadIdx.x;
    if (t < 128) l[t] = lat[(size_t)b * 128 + t];
    __syncthreads();

    { // deconv1 -> (64,2,2), one output per thread
        int o = t >> 2, y = (t >> 1) & 1, x = t & 1;
        const float* wo = dw1 + (size_t)o * 128 * 16 + (2 - y) * 4 + (2 - x);
        float acc = db1[o];
        for (int i = 0; i < 128; ++i) acc = fmaf(l[i], wo[i * 16], acc);
        d1s[t] = fmaxf(acc, 0.0f);
    }
    __syncthreads();

    // deconv2 -> (32,4,4): 512 outputs, 2 per thread
#pragma unroll
    for (int u = 0; u < 2; ++u) {
        int oi = t + u * 256;                  // o*16 + y*4 + x
        int o = oi >> 4, y = (oi >> 2) & 3, x = oi & 3;
        float acc = db2[o];
#pragma unroll
        for (int iy = 0; iy < 2; ++iy) {
            int ky = 2 + 2 * iy - y; if ((unsigned)ky > 3u) continue;
#pragma unroll
            for (int ix = 0; ix < 2; ++ix) {
                int kx = 2 + 2 * ix - x; if ((unsigned)kx > 3u) continue;
                const float* wo = dw2 + (size_t)o * 64 * 16 + ky * 4 + kx;
                const float* din = d1s + iy * 2 + ix;
                for (int i = 0; i < 64; ++i) acc = fmaf(din[i * 4], wo[i * 16], acc);
            }
        }
        d2s[oi] = fmaxf(acc, 0.0f);
    }
    __syncthreads();

    if (t < 64) {                              // deconv3 -> (1,8,8)
        int y = t >> 3, x = t & 7;
        float acc = db3[0];
#pragma unroll
        for (int iy = 0; iy < 4; ++iy) {
            int ky = 2 + 2 * iy - y; if ((unsigned)ky > 3u) continue;
#pragma unroll
            for (int ix = 0; ix < 4; ++ix) {
                int kx = 2 + 2 * ix - x; if ((unsigned)kx > 3u) continue;
                const float* wo = dw3 + ky * 4 + kx;   // dw3: (1,32,4,4)
                const float* din = d2s + iy * 4 + ix;
                for (int i = 0; i < 32; ++i) acc = fmaf(din[i * 16], wo[i * 16], acc);
            }
        }
        recon[(size_t)b * 64 + t] = acc;
    }
}

// ---------------------------------------------------------------------------
extern "C" void kernel_launch(void* const* d_in, const int* in_sizes, int n_in,
                              void* d_out, int out_size, void* d_ws, size_t ws_size,
                              hipStream_t stream) {
    (void)in_sizes; (void)n_in; (void)out_size; (void)ws_size;

    const float* x     = (const float*)d_in[0];
    const float* w1    = (const float*)d_in[1];
    const float* b1    = (const float*)d_in[2];
    const float* w2    = (const float*)d_in[3];
    const float* b2    = (const float*)d_in[4];
    const float* w3    = (const float*)d_in[5];
    const float* b3    = (const float*)d_in[6];
    const float* w_h2l = (const float*)d_in[7];
    const float* b_h2l = (const float*)d_in[8];
    const float* emb   = (const float*)d_in[9];
    const float* w_p2h = (const float*)d_in[10];
    const float* b_p2h = (const float*)d_in[11];
    const float* dw1   = (const float*)d_in[12];
    const float* db1   = (const float*)d_in[13];
    const float* dw2   = (const float*)d_in[14];
    const float* db2   = (const float*)d_in[15];
    const float* dw3   = (const float*)d_in[16];
    const float* db3   = (const float*)d_in[17];

    float* out   = (float*)d_out;          // recon | z_e | z_q | vq_loss
    float* recon = out;                    // 256*1*8*8   = 16384
    float* z_e   = out + 16384;            // 256*256     = 65536
    float* z_q   = out + 81920;            // 256*256     = 65536
    float* loss  = out + 147456;           // 1

    float* ws    = (float*)d_ws;
    float* h1    = ws;                                  // 256*32*64*64
    float* h2    = h1 + (size_t)256 * 32 * 4096;        // 256*64*32*32
    float* h3    = h2 + (size_t)256 * 64 * 1024;        // 256*128*16*16
    float* feat  = h3 + (size_t)256 * 128 * 256;        // 256*128
    float* lat   = feat + 32768;                        // 256*128
    float* ploss = lat + 32768;                         // 256

    // Dynamic LDS bytes: f16 weight slab + fp32 halo tile.
    const size_t smem2 = (size_t)16 * 512 * 2 + (size_t)32 * 10 * 72 * 4;   // 108544
    const size_t smem3 = (size_t)16 * 1024 * 2 + (size_t)64 * 18 * 40 * 4;  // 217088

    conv1_kernel<<<4096, 256, 0, stream>>>(x, w1, b1, h1);
    conv_wmma_kernel<32, 64, 64, 32><<<dim3(2048, 4), 256, smem2, stream>>>(h1, w2, b2, h2);
    conv_wmma_kernel<64, 128, 32, 16><<<dim3(512, 8), 256, smem3, stream>>>(h2, w3, b3, h3);
    pool_kernel<<<128, 256, 0, stream>>>(h3, feat);
    gemm_nt_wmma<<<dim3(16, 16), 32, 0, stream>>>(feat, w_h2l, b_h2l, z_e, 256, 128);
    vq_kernel<<<256, 256, 0, stream>>>(z_e, emb, z_q, ploss);
    loss_reduce<<<1, 256, 0, stream>>>(ploss, loss);
    gemm_nt_wmma<<<dim3(8, 16), 32, 0, stream>>>(z_q, w_p2h, b_p2h, lat, 128, 256);
    decoder_kernel<<<256, 256, 0, stream>>>(lat, dw1, db1, dw2, db2, dw3, db3, recon);
}
